// Net_89575837925978
// MI455X (gfx1250) — matrix operand order repaired and checked by reading, not compiled, not run on hardware
//
#include <hip/hip_runtime.h>
#include <hip/hip_bf16.h>
#include <math.h>

#define GN 50000
#define GE 800000
#define GH 4
#define GD 64
#define GF 256   // H*D
#define NEG_SLOPE 0.2f

typedef _Float16 h16 __attribute__((ext_vector_type(16)));
typedef float    f8  __attribute__((ext_vector_type(8)));

// ---------- helpers ----------
__device__ __forceinline__ float lrelu(float x) { return x > 0.f ? x : NEG_SLOPE * x; }

__device__ __forceinline__ unsigned enc_max(float x) {
    int i = __float_as_int(x);
    return (i >= 0) ? ((unsigned)i | 0x80000000u) : ~(unsigned)i;
}
__device__ __forceinline__ float dec_max(unsigned k) {
    if (k == 0u) return 0.f;  // empty segment -> 0 (matches reference guard)
    int i = (k & 0x80000000u) ? (int)(k ^ 0x80000000u) : (int)~k;
    return __int_as_float(i);
}

// ---------- layer 1: feat = pos@W1 (K=2), out(residual) = pos@resW1 ----------
__global__ __launch_bounds__(256)
void k_layer1(const float* __restrict__ pos, const float* __restrict__ W1,
              const float* __restrict__ resW1, float* __restrict__ feat,
              float* __restrict__ out) {
    int t = blockIdx.x * blockDim.x + threadIdx.x;
    if (t >= GN * GF) return;
    int n = t >> 8, c = t & 255;
    float p0 = pos[(size_t)n * 2 + 0], p1 = pos[(size_t)n * 2 + 1];
    feat[t] = p0 * W1[c]    + p1 * W1[GF + c];
    out[t]  = p0 * resW1[c] + p1 * resW1[GF + c];
}

// ---------- transpose + cvt f32 W[K][Nn] -> f16 Wt[Nn][K] ----------
__global__ __launch_bounds__(256)
void k_transpose_cvt(const float* __restrict__ W, _Float16* __restrict__ Wt,
                     int K, int Nn) {
    int t = blockIdx.x * blockDim.x + threadIdx.x;
    if (t >= K * Nn) return;
    int k = t / Nn, n = t % Nn;
    Wt[(size_t)n * K + k] = (_Float16)W[(size_t)k * Nn + n];
}

// ---------- WMMA GEMM: one wave computes a full 16x256 output strip ----------
// C[rows,256] = A[rows,256](f32->f16 on the fly) x W (pre-transposed f16 Bt[n][k])
// A fragment is built ONCE per k-step and reused across all 16 column tiles.
__global__ __launch_bounds__(32)
void k_gemm256_wmma(const float* __restrict__ A, const _Float16* __restrict__ Bt,
                    float* __restrict__ C) {
    int tm = blockIdx.x;                 // row tile (rows/16)
    int lane = threadIdx.x;
    int r = lane & 15, g = lane >> 4;
    const float*    Ar  = A  + (size_t)(tm * 16 + r) * GF;
    const _Float16* Br0 = Bt + (size_t)r * GF;
    f8 acc[16];
    #pragma unroll
    for (int t = 0; t < 16; ++t) acc[t] = (f8){};
    for (int k0 = 0; k0 < GF; k0 += 32) {
        h16 a;
        const float* a0 = Ar + k0 + g * 8;
        const float* a1 = Ar + k0 + 16 + g * 8;
        #pragma unroll
        for (int i = 0; i < 8; ++i) {
            a[i]     = (_Float16)a0[i];
            a[i + 8] = (_Float16)a1[i];
        }
        #pragma unroll
        for (int tn = 0; tn < 16; ++tn) {    // 16 independent WMMAs per k-step
            h16 b; { typedef _Float16 h8v __attribute__((ext_vector_type(8))); const _Float16* bp = Br0 + (size_t)tn * 16 * GF + k0 + g * 8; h8v b0 = *(const h8v*)bp; h8v b1 = *(const h8v*)(bp + 16); b = __builtin_shufflevector(b0, b1, 0,1,2,3,4,5,6,7,8,9,10,11,12,13,14,15); }
            acc[tn] = __builtin_amdgcn_wmma_f32_16x16x32_f16(false, a, false, b,
                                                             (short)0, acc[tn],
                                                             false, false);
        }
    }
    #pragma unroll
    for (int tn = 0; tn < 16; ++tn) {
        float* Cp = C + (size_t)(tm * 16) * GF + tn * 16 + r;
        #pragma unroll
        for (int j = 0; j < 8; ++j)
            Cp[(size_t)(j + g * 8) * GF] = acc[tn][j];
    }
}

// ---------- attention logits el/er per (node, head) ----------
__global__ __launch_bounds__(256)
void k_el_er(const float* __restrict__ feat, const float* __restrict__ al,
             const float* __restrict__ ar, float* __restrict__ el,
             float* __restrict__ er) {
    int t = blockIdx.x * blockDim.x + threadIdx.x;
    if (t >= GN * GH) return;
    int n = t >> 2, h = t & 3;
    const float* f = feat + (size_t)n * GF + h * GD;
    const float* a = al + h * GD;
    const float* b = ar + h * GD;
    float sl = 0.f, sr = 0.f;
    #pragma unroll 4
    for (int i = 0; i < GD; ++i) { sl += f[i] * a[i]; sr += f[i] * b[i]; }
    el[t] = sl; er[t] = sr;
}

// ---------- edge softmax pass 1: segment max via int-keyed atomicMax ----------
__global__ __launch_bounds__(256)
void k_edge_max(const float* __restrict__ el, const float* __restrict__ er,
                const int* __restrict__ src, const int* __restrict__ dst,
                unsigned* __restrict__ emax) {
    int e = blockIdx.x * blockDim.x + threadIdx.x;
    if (e >= GE) return;
    int s = src[e], d = dst[e];
    #pragma unroll
    for (int h = 0; h < GH; ++h) {
        float ev = lrelu(el[s * GH + h] + er[d * GH + h]);
        atomicMax(&emax[d * GH + h], enc_max(ev));
    }
}

// ---------- edge softmax pass 2: denominator ----------
__global__ __launch_bounds__(256)
void k_edge_den(const float* __restrict__ el, const float* __restrict__ er,
                const int* __restrict__ src, const int* __restrict__ dst,
                const unsigned* __restrict__ emax, float* __restrict__ den) {
    int e = blockIdx.x * blockDim.x + threadIdx.x;
    if (e >= GE) return;
    int s = src[e], d = dst[e];
    #pragma unroll
    for (int h = 0; h < GH; ++h) {
        float ev = lrelu(el[s * GH + h] + er[d * GH + h]);
        atomicAdd(&den[d * GH + h], expf(ev - dec_max(emax[d * GH + h])));
    }
}

// ---------- edge aggregate: out[dst] += feat[src] * alpha  (thread per edge-head) ----------
__global__ __launch_bounds__(256)
void k_edge_aggr(const float* __restrict__ feat, const float* __restrict__ el,
                 const float* __restrict__ er, const int* __restrict__ src,
                 const int* __restrict__ dst, const unsigned* __restrict__ emax,
                 const float* __restrict__ den, float* __restrict__ out) {
    int t = blockIdx.x * blockDim.x + threadIdx.x;
    if (t >= GE * GH) return;
    int e = t >> 2, h = t & 3;
    int s = src[e], d = dst[e];
    float ev = lrelu(el[s * GH + h] + er[d * GH + h]);
    float ex = expf(ev - dec_max(emax[d * GH + h]));
    float alpha = ex / fmaxf(den[d * GH + h], 1e-9f);
    const float* fs = feat + (size_t)s * GF + h * GD;
    float*       od = out  + (size_t)d * GF + h * GD;
    #pragma unroll 4
    for (int i = 0; i < GD; ++i) atomicAdd(&od[i], fs[i] * alpha);
}

// ---------- in-place relu ----------
__global__ __launch_bounds__(256)
void k_relu(float* __restrict__ x, int n) {
    int t = blockIdx.x * blockDim.x + threadIdx.x;
    if (t < n) x[t] = fmaxf(x[t], 0.f);
}

// ---------- head mean: [N,H,D] -> [N,D] ----------
__global__ __launch_bounds__(256)
void k_mean(const float* __restrict__ x, float* __restrict__ hm) {
    int t = blockIdx.x * blockDim.x + threadIdx.x;
    if (t >= GN * GD) return;
    int n = t >> 6, d = t & 63;
    const float* p = x + (size_t)n * GF + d;
    hm[t] = 0.25f * (p[0] + p[GD] + p[2 * GD] + p[3 * GD]);
}

// ---------- edge MLP: wave per 16 edges, WMMA 16x64 hidden, fused ReLU + mW2 dot + sigmoid ----------
__global__ __launch_bounds__(32)
void k_edge_mlp(const float* __restrict__ hm, const int* __restrict__ src,
                const int* __restrict__ dst, const _Float16* __restrict__ mW1t,
                const float* __restrict__ mb1, const float* __restrict__ mW2,
                const float* __restrict__ mb2, float* __restrict__ out) {
    __shared__ _Float16 xs[16 * 64];
    __shared__ float rowacc[16];
    int lane = threadIdx.x;
    int e0 = blockIdx.x * 16;
    // build x = |hm[src]-hm[dst]| for 16 edges into LDS (f16)
    #pragma unroll
    for (int i = 0; i < 32; ++i) {
        int idx = lane + i * 32;
        int row = idx >> 6, d = idx & 63;
        int e = e0 + row;
        float xv = fabsf(hm[(size_t)src[e] * GD + d] - hm[(size_t)dst[e] * GD + d]);
        xs[idx] = (_Float16)xv;
    }
    if (lane < 16) rowacc[lane] = 0.f;
    __syncthreads();
    int r = lane & 15, g = lane >> 4;
    float rowpart[8] = {};
    #pragma unroll
    for (int n0 = 0; n0 < 4; ++n0) {               // 4 col tiles of 16 -> 64 hidden
        f8 acc = {};
        const _Float16* Br = mW1t + (size_t)(n0 * 16 + r) * GD;
        #pragma unroll
        for (int k0 = 0; k0 < GD; k0 += 32) {
            h16 a, b;
            const _Float16* a0 = &xs[r * GD + k0 + g * 8];
            const _Float16* a1 = &xs[r * GD + k0 + 16 + g * 8];
            #pragma unroll
            for (int i = 0; i < 8; ++i) { a[i] = a0[i]; a[i + 8] = a1[i]; }
            { typedef _Float16 h8v __attribute__((ext_vector_type(8))); h8v b0 = *(const h8v*)(Br + k0 + g * 8); h8v b1 = *(const h8v*)(Br + k0 + 16 + g * 8); b = __builtin_shufflevector(b0, b1, 0,1,2,3,4,5,6,7,8,9,10,11,12,13,14,15); }
            acc = __builtin_amdgcn_wmma_f32_16x16x32_f16(false, a, false, b,
                                                         (short)0, acc, false, false);
        }
        int col = n0 * 16 + r;
        float b1 = mb1[col], w2 = mW2[col];
        #pragma unroll
        for (int j = 0; j < 8; ++j) {
            float hc = fmaxf(acc[j] + b1, 0.f);    // Linear+ReLU
            rowpart[j] += hc * w2;                 // partial of hidden @ mW2
        }
    }
    #pragma unroll
    for (int j = 0; j < 8; ++j) atomicAdd(&rowacc[j + g * 8], rowpart[j]);
    __syncthreads();
    if (lane < 16) {
        float s = rowacc[lane] + mb2[0];
        out[e0 + lane] = 1.f / (1.f + expf(-s));   // sigmoid
    }
}

// =====================================================================
extern "C" void kernel_launch(void* const* d_in, const int* in_sizes, int n_in,
                              void* d_out, int out_size, void* d_ws, size_t ws_size,
                              hipStream_t stream) {
    (void)in_sizes; (void)n_in; (void)out_size; (void)ws_size;
    const float* pos   = (const float*)d_in[0];
    const int*   src   = (const int*)  d_in[1];
    const int*   dst   = (const int*)  d_in[2];
    const float* W1    = (const float*)d_in[3];
    const float* al1   = (const float*)d_in[4];
    const float* ar1   = (const float*)d_in[5];
    const float* resW1 = (const float*)d_in[6];
    const float* W2    = (const float*)d_in[7];
    const float* al2   = (const float*)d_in[8];
    const float* ar2   = (const float*)d_in[9];
    const float* W3    = (const float*)d_in[10];
    const float* al3   = (const float*)d_in[11];
    const float* ar3   = (const float*)d_in[12];
    const float* mW1   = (const float*)d_in[13];
    const float* mb1   = (const float*)d_in[14];
    const float* mW2   = (const float*)d_in[15];
    const float* mb2   = (const float*)d_in[16];
    float* score = (float*)d_out;

    // workspace carve-up (256B aligned)
    size_t off = 0;
    auto carve = [&](size_t bytes) {
        void* p = (char*)d_ws + off;
        off += (bytes + 255) & ~(size_t)255;
        return p;
    };
    float*     hA    = (float*)carve((size_t)GN * GF * 4);
    float*     hB    = (float*)carve((size_t)GN * GF * 4);
    float*     Ft    = (float*)carve((size_t)GN * GF * 4);
    float*     el    = (float*)carve((size_t)GN * GH * 4);
    float*     er    = (float*)carve((size_t)GN * GH * 4);
    unsigned*  emax  = (unsigned*)carve((size_t)GN * GH * 4);
    float*     den   = (float*)carve((size_t)GN * GH * 4);
    float*     hmean = (float*)carve((size_t)GN * GD * 4);
    _Float16*  Wt    = (_Float16*)carve((size_t)GF * GF * 2);
    _Float16*  mW1t  = (_Float16*)carve((size_t)GD * GD * 2);

    const int BLK = 256;
    const int gNF   = (GN * GF + BLK - 1) / BLK;
    const int gNH   = (GN * GH + BLK - 1) / BLK;
    const int gE    = (GE + BLK - 1) / BLK;
    const int gEH   = (GE * GH + BLK - 1) / BLK;
    const int gND   = (GN * GD + BLK - 1) / BLK;
    const int gGEMM = GN / 16;          // one wave per 16-row strip (all 16 col tiles)
    const size_t featBytes = (size_t)GN * GF * 4;

    auto edge_softmax = [&](const float* feat, float* out) {
        hipMemsetAsync(emax, 0, (size_t)GN * GH * 4, stream);
        hipMemsetAsync(den,  0, (size_t)GN * GH * 4, stream);
        k_edge_max <<<gE,  BLK, 0, stream>>>(el, er, src, dst, emax);
        k_edge_den <<<gE,  BLK, 0, stream>>>(el, er, src, dst, emax, den);
        k_edge_aggr<<<gEH, BLK, 0, stream>>>(feat, el, er, src, dst, emax, den, out);
    };

    // ---- layer 1 (K=2 projections scalar; residual = pos@resW1) ----
    k_layer1<<<gNF, BLK, 0, stream>>>(pos, W1, resW1, Ft, hB);
    k_el_er <<<gNH, BLK, 0, stream>>>(Ft, al1, ar1, el, er);
    edge_softmax(Ft, hB);
    k_relu<<<gNF, BLK, 0, stream>>>(hB, GN * GF);

    // ---- layer 2 (WMMA GEMM, identity residual) ----
    k_transpose_cvt<<<(GF * GF + BLK - 1) / BLK, BLK, 0, stream>>>(W2, Wt, GF, GF);
    k_gemm256_wmma <<<gGEMM, 32, 0, stream>>>(hB, Wt, Ft);
    k_el_er        <<<gNH, BLK, 0, stream>>>(Ft, al2, ar2, el, er);
    hipMemcpyAsync(hA, hB, featBytes, hipMemcpyDeviceToDevice, stream);
    edge_softmax(Ft, hA);
    k_relu<<<gNF, BLK, 0, stream>>>(hA, GN * GF);

    // ---- layer 3 (WMMA GEMM, identity residual, no relu, head mean) ----
    k_transpose_cvt<<<(GF * GF + BLK - 1) / BLK, BLK, 0, stream>>>(W3, Wt, GF, GF);
    k_gemm256_wmma <<<gGEMM, 32, 0, stream>>>(hA, Wt, Ft);
    k_el_er        <<<gNH, BLK, 0, stream>>>(Ft, al3, ar3, el, er);
    hipMemcpyAsync(hB, hA, featBytes, hipMemcpyDeviceToDevice, stream);
    edge_softmax(Ft, hB);
    k_mean<<<gND, BLK, 0, stream>>>(hB, hmean);

    // ---- edge-scoring MLP (WMMA, wave per 16 edges) ----
    k_transpose_cvt<<<(GD * GD + BLK - 1) / BLK, BLK, 0, stream>>>(mW1, mW1t, GD, GD);
    k_edge_mlp<<<GE / 16, 32, 0, stream>>>(hmean, src, dst, mW1t, mb1, mW2, mb2, score);
}
